// StudentBoWCLF_3547642986555
// MI455X (gfx1250) — compile-verified, matrix operand order checked by reading, not run on hardware
//
#include <hip/hip_runtime.h>
#include <cstdint>

#define VOCAB 50000
#define NUM_CLASSES 12
#define BATCH 4096
#define SEQ 200

typedef __attribute__((ext_vector_type(16))) _Float16 v16h;
typedef __attribute__((ext_vector_type(8)))  float    v8f;

// Kernel 1: repack W[12][50000] (row-major) into Wt[v][16]: each entry is an
// exact two-term f16 split of the f32 weight packed in one dword
// (lo16 = f16(w), hi16 = f16(w - f16(w))). Classes 12..15 zero-padded.
// One vocab entry = 64B = one cache line.
__global__ void bow_prep_kernel(const float* __restrict__ W,
                                uint32_t* __restrict__ Wt) {
  int t = blockIdx.x * blockDim.x + threadIdx.x;
  if (t >= VOCAB * 16) return;
  int c = t / VOCAB;          // 0..15, reads coalesced over v
  int v = t - c * VOCAB;
  float val = (c < NUM_CLASSES) ? W[c * VOCAB + v] : 0.0f;
  union { _Float16 h; unsigned short u; } hi, lo;
  hi.h = (_Float16)val;
  lo.h = (_Float16)(val - (float)hi.h);
  Wt[(uint32_t)v * 16u + (uint32_t)c] = ((uint32_t)lo.u << 16) | (uint32_t)hi.u;
}

// Kernel 2: one wave32 per batch row.
//   logits[c] = bias[c] + sum_s (hi + lo)(W[c, ids[s]])
// via D = ones(16x32,f16) x B(32x16,f16) + C (f32 accumulate). A == ones means
// D[m,n] = sum_k B[k,n], so K-slot assignment is free: lane l loads 8 packed
// (hi,lo) dwords for class n = l%16, covering 8 seq positions per chunk.
// Two independent accumulator chains (even/odd chunks) for memory parallelism.
__global__ __launch_bounds__(256) void bow_logits_kernel(
    const int* __restrict__ ids, const uint32_t* __restrict__ Wt,
    const float* __restrict__ bias, float* __restrict__ out) {
  const int lane = threadIdx.x & 31;
  const int wave = threadIdx.x >> 5;
  const int row  = blockIdx.x * (blockDim.x >> 5) + wave;  // grid covers 4096 exactly
  const int h = lane >> 4;          // which 8-position sub-block of each 16-chunk
  const uint32_t c = (uint32_t)(lane & 15);  // class handled by this lane
  const int* __restrict__ rid = ids + (size_t)row * SEQ;

  v16h ones;
#pragma unroll
  for (int i = 0; i < 16; ++i) ones[i] = (_Float16)1.0f;

  v8f acc0 = {0.f, 0.f, 0.f, 0.f, 0.f, 0.f, 0.f, 0.f};
  v8f acc1 = {0.f, 0.f, 0.f, 0.f, 0.f, 0.f, 0.f, 0.f};

  // Gather helper: unsigned 32-bit dword index -> saddr + voffset form.
  auto gat = [&](int id) -> uint32_t {
    return Wt[(((uint32_t)id) << 4) + c];
  };

  // 12 full chunks of 16 positions, processed in pairs on independent chains.
#pragma unroll
  for (int p = 0; p < 6; ++p) {
    const int base0 = (2 * p) * 16 + h * 8;       // 32B-aligned
    const int base1 = (2 * p + 1) * 16 + h * 8;
    int4 a0 = *reinterpret_cast<const int4*>(rid + base0);
    int4 a1 = *reinterpret_cast<const int4*>(rid + base0 + 4);
    int4 b0 = *reinterpret_cast<const int4*>(rid + base1);
    int4 b1 = *reinterpret_cast<const int4*>(rid + base1 + 4);
    union { v16h v; uint32_t u[8]; } B0, B1;
    B0.u[0] = gat(a0.x);  B0.u[1] = gat(a0.y);
    B0.u[2] = gat(a0.z);  B0.u[3] = gat(a0.w);
    B0.u[4] = gat(a1.x);  B0.u[5] = gat(a1.y);
    B0.u[6] = gat(a1.z);  B0.u[7] = gat(a1.w);
    B1.u[0] = gat(b0.x);  B1.u[1] = gat(b0.y);
    B1.u[2] = gat(b0.z);  B1.u[3] = gat(b0.w);
    B1.u[4] = gat(b1.x);  B1.u[5] = gat(b1.y);
    B1.u[6] = gat(b1.z);  B1.u[7] = gat(b1.w);
    acc0 = __builtin_amdgcn_wmma_f32_16x16x32_f16(false, ones, false, B0.v,
                                                  (short)0, acc0, false, false);
    acc1 = __builtin_amdgcn_wmma_f32_16x16x32_f16(false, ones, false, B1.v,
                                                  (short)0, acc1, false, false);
  }

  // Tail: positions 192..199 live in the h==0 half; h==1 contributes zeros.
  {
    union { v16h v; uint32_t u[8]; } B;
    if (h == 0) {
      int4 a0 = *reinterpret_cast<const int4*>(rid + 192);
      int4 a1 = *reinterpret_cast<const int4*>(rid + 196);
      B.u[0] = gat(a0.x);  B.u[1] = gat(a0.y);
      B.u[2] = gat(a0.z);  B.u[3] = gat(a0.w);
      B.u[4] = gat(a1.x);  B.u[5] = gat(a1.y);
      B.u[6] = gat(a1.z);  B.u[7] = gat(a1.w);
    } else {
#pragma unroll
      for (int j = 0; j < 8; ++j) B.u[j] = 0u;
    }
    // Reconverged: EXEC all ones at the WMMA.
    acc0 = __builtin_amdgcn_wmma_f32_16x16x32_f16(false, ones, false, B.v,
                                                  (short)0, acc0, false, false);
  }

  // Every D row is identical -> lane l holds class (l%16) total in element 0.
  float total = acc0[0] + acc1[0];
  float x = (c < NUM_CLASSES) ? (total + bias[c]) : -1e30f;

  // log_softmax across the 16-lane half (classes 12..15 masked out).
  float m = x;
#pragma unroll
  for (int off = 1; off < 16; off <<= 1)
    m = fmaxf(m, __shfl_xor(m, off, 32));
  float e = expf(x - m);
  float s = e;
#pragma unroll
  for (int off = 1; off < 16; off <<= 1)
    s += __shfl_xor(s, off, 32);
  float res = x - m - logf(s);

  if (h == 0 && c < NUM_CLASSES)
    out[(size_t)row * NUM_CLASSES + c] = res;
}

extern "C" void kernel_launch(void* const* d_in, const int* in_sizes, int n_in,
                              void* d_out, int out_size, void* d_ws, size_t ws_size,
                              hipStream_t stream) {
  const int*   ids  = (const int*)d_in[0];
  const float* W    = (const float*)d_in[1];
  const float* bias = (const float*)d_in[2];
  float*       out  = (float*)d_out;
  uint32_t*    Wt   = (uint32_t*)d_ws;   // 50000*16*4 = 3.2 MB

  (void)in_sizes; (void)n_in; (void)out_size; (void)ws_size;

  const int prepN = VOCAB * 16;
  bow_prep_kernel<<<(prepN + 255) / 256, 256, 0, stream>>>(W, Wt);

  // 8 waves (rows) per 256-thread block -> 512 blocks cover 4096 rows exactly.
  bow_logits_kernel<<<BATCH / 8, 256, 0, stream>>>(ids, Wt, bias, out);
}
